// Model_8022998909298
// MI455X (gfx1250) — compile-verified, matrix operand order checked by reading, not run on hardware
//
#include <hip/hip_runtime.h>
#include <hip/hip_bf16.h>
#include <math.h>

typedef __attribute__((ext_vector_type(16))) _Float16 v16h;
typedef __attribute__((ext_vector_type(8)))  float    v8f;

#define MAXDEG 256           // Poisson(64) over 20k nodes: max deg ~110; 256 is safe
#define TROW   12            // table row stride (floats): 48B, 16B-aligned rows
#define INV_SQRT3F 0.57735026918962576f

__device__ __forceinline__ float gelu_tanh(float x) {
    // jax.nn.gelu default (approximate=True, tanh form)
    float x3 = x * x * x;
    return 0.5f * x * (1.0f + tanhf(0.7978845608028654f * (x + 0.044715f * x3)));
}

// ---------------------------------------------------------------------------
// K1: row_start[i] = lower_bound(src, i) over sorted src; zero `pooled`;
//     prepack rec[2i] = (pos.xyz, 0) so later passes gather float4s.
// ---------------------------------------------------------------------------
__global__ void k_rowstart(const int* __restrict__ src, const float* __restrict__ pos,
                           int E, int n,
                           int* __restrict__ row_start, float4* __restrict__ rec,
                           float* __restrict__ pooled) {
    int i = blockIdx.x * blockDim.x + threadIdx.x;
    if (i < 10) pooled[i] = 0.0f;
    if (i > n) return;
    if (i < n)
        rec[2 * i] = make_float4(pos[3 * i + 0], pos[3 * i + 1], pos[3 * i + 2], 0.0f);
    if (i == n) { row_start[n] = E; return; }
    int lo = 0, hi = E;
    while (lo < hi) {
        int mid = (lo + hi) >> 1;
        if (src[mid] < i) lo = mid + 1; else hi = mid;
    }
    row_start[i] = lo;
}

// ---------------------------------------------------------------------------
// K2: one wave32 per node i. S[i] = sum of rhat over INCOMING edges =
// -(sum of outgoing rhat) by graph symmetry; deg[i] = segment length.
// Writes the packed 32B node record: rec[2i]=(pos,deg), rec[2i+1]=(S,0).
// (Rewriting rec[2i].xyz with identical values is a benign dword-level race.)
// ---------------------------------------------------------------------------
__global__ void k_sdeg(const float4* __restrict__ recR, const int* __restrict__ dst,
                       const int* __restrict__ row_start, int n,
                       float4* __restrict__ rec) {
    int wave = (blockIdx.x * (blockDim.x >> 5)) + (threadIdx.x >> 5);
    int lane = threadIdx.x & 31;
    if (wave >= n) return;
    int s = row_start[wave], e = row_start[wave + 1];
    float4 pi = recR[2 * wave];
    float sx = 0.f, sy = 0.f, sz = 0.f;
    for (int t = s + lane; t < e; t += 32) {
        __builtin_prefetch(dst + t + 32, 0, 0);
        int d = dst[t];
        float4 pd = recR[2 * d];
        float rx = pd.x - pi.x, ry = pd.y - pi.y, rz = pd.z - pi.z;
        float inv = rsqrtf(rx * rx + ry * ry + rz * rz);
        sx += rx * inv; sy += ry * inv; sz += rz * inv;
    }
    #pragma unroll
    for (int off = 16; off >= 1; off >>= 1) {
        sx += __shfl_xor(sx, off, 32);
        sy += __shfl_xor(sy, off, 32);
        sz += __shfl_xor(sz, off, 32);
    }
    if (lane == 0) {
        rec[2 * wave]     = make_float4(pi.x, pi.y, pi.z, (float)(e - s));
        rec[2 * wave + 1] = make_float4(-sx, -sy, -sz, 0.0f);
    }
}

// ---------------------------------------------------------------------------
// K3: a_table[d][j] = sum_k gelu(d*W1_0e[k]) * W2_0e[k][j], d = 0..255,
// via V_WMMA_F32_16X16X32_F16 (16 degrees per wave; K=5 zero-padded to 32,
// N=10 zero-padded to 16). Rows stored with stride TROW=12 (16B-aligned).
// Also c2 = W1_1o @ W2_1o. Launch: 1 block x 512 threads (16 waves);
// EXEC all-1s at the WMMA.
// ---------------------------------------------------------------------------
__global__ void k_table(const float* __restrict__ W1_0e, const float* __restrict__ W2_0e,
                        const float* __restrict__ W1_1o, const float* __restrict__ W2_1o,
                        float* __restrict__ a_table, float* __restrict__ c2) {
    int lane = threadIdx.x & 31;
    int wid  = threadIdx.x >> 5;      // 0..15
    int base = wid * 16;              // first degree of this tile

    v16h af = {};                     // A 16x32 f16: lanes 0-15, half h = K=h (h<8)
    v16h bf = {};                     // B 32x16 f16: lanes 0-15, half h = K=h, N=lane
    v8f  c  = {};
    if (lane < 16) {
        float d = (float)(base + lane);           // row M = lane -> degree
        #pragma unroll
        for (int k = 0; k < 5; ++k)
            af[k] = (_Float16)gelu_tanh(d * W1_0e[k]);
        if (lane < 10) {
            #pragma unroll
            for (int k = 0; k < 5; ++k)
                bf[k] = (_Float16)W2_0e[k * 10 + lane];
        }
    }
    c = __builtin_amdgcn_wmma_f32_16x16x32_f16(false, af, false, bf,
                                               (short)0, c, false, false);
    // D layout: VGPR r: lanes 0-15 -> (M=r, N=lane); lanes 16-31 -> (M=r+8, N=lane-16)
    int col  = lane & 15;
    int moff = (lane < 16) ? 0 : 8;
    if (col < 10) {
        #pragma unroll
        for (int r = 0; r < 8; ++r)
            a_table[(base + r + moff) * TROW + col] = c[r];
    } else if (col < 12) {
        #pragma unroll
        for (int r = 0; r < 8; ++r)               // zero the 2 pad columns
            a_table[(base + r + moff) * TROW + col] = 0.0f;
    }
    if (threadIdx.x < 10) {
        float acc = 0.f;
        #pragma unroll
        for (int k = 0; k < 5; ++k) acc += W1_1o[k] * W2_1o[k * 10 + threadIdx.x];
        c2[threadIdx.x] = acc;
    }
}

// ---------------------------------------------------------------------------
// K4: one wave32 per node i (segment by src):
//   A[j] = sum_e a_table[deg(dst_e)][j];  b = sum_e S[dst_e].(-rhat_e)
//   h[j] = gelu(A[j] + b*INV_SQRT3*c2[j]);  pooled[j] += h[j]
// 1024-thread blocks (32 nodes) amortize the 12KB LDS table stage; per-edge
// gathers are two adjacent b128 loads (one 32B record); table reads are
// 3x ds_load_b128 per edge (stride-12 rows).
// ---------------------------------------------------------------------------
__global__ void __launch_bounds__(1024)
k_accum(const float4* __restrict__ rec, const int* __restrict__ dst,
        const int* __restrict__ row_start,
        const float4* __restrict__ a_table_g4, const float* __restrict__ c2_g,
        int n, float* __restrict__ pooled) {
    __shared__ float4 tab4[MAXDEG * (TROW / 4)];
    __shared__ float  c2s[10];
    for (int t = threadIdx.x; t < MAXDEG * (TROW / 4); t += blockDim.x)
        tab4[t] = a_table_g4[t];
    if (threadIdx.x < 10) c2s[threadIdx.x] = c2_g[threadIdx.x];
    __syncthreads();

    int wave = (blockIdx.x * (blockDim.x >> 5)) + (threadIdx.x >> 5);
    int lane = threadIdx.x & 31;
    if (wave >= n) return;

    int s = row_start[wave], e = row_start[wave + 1];
    float4 pi = rec[2 * wave];
    float acc[10] = {0.f, 0.f, 0.f, 0.f, 0.f, 0.f, 0.f, 0.f, 0.f, 0.f};
    float accb = 0.f;

    for (int t = s + lane; t < e; t += 32) {
        __builtin_prefetch(dst + t + 32, 0, 0);
        int d = dst[t];
        float4 pd = rec[2 * d];        // pos.xyz, deg
        float4 sd = rec[2 * d + 1];    // S.xyz
        float rx = pd.x - pi.x, ry = pd.y - pi.y, rz = pd.z - pi.z;
        float inv = rsqrtf(rx * rx + ry * ry + rz * rz);
        // incoming-edge rhat at this node is -(outgoing rhat)
        accb -= (sd.x * rx + sd.y * ry + sd.z * rz) * inv;
        int dg = (int)pd.w;
        if (dg > MAXDEG - 1) dg = MAXDEG - 1;
        const float4* tr = &tab4[dg * (TROW / 4)];
        float4 t0 = tr[0], t1 = tr[1], t2 = tr[2];
        acc[0] += t0.x; acc[1] += t0.y; acc[2] += t0.z; acc[3] += t0.w;
        acc[4] += t1.x; acc[5] += t1.y; acc[6] += t1.z; acc[7] += t1.w;
        acc[8] += t2.x; acc[9] += t2.y;
    }
    #pragma unroll
    for (int off = 16; off >= 1; off >>= 1) {
        accb += __shfl_xor(accb, off, 32);
        #pragma unroll
        for (int j = 0; j < 10; ++j) acc[j] += __shfl_xor(acc[j], off, 32);
    }
    if (lane < 10) {
        float h = gelu_tanh(acc[lane] + accb * INV_SQRT3F * c2s[lane]);
        atomicAdd(&pooled[lane], h);
    }
}

// ---------------------------------------------------------------------------
// K5: out = softmax(mean(h) @ mlp_w + mlp_b). One wave.
// ---------------------------------------------------------------------------
__global__ void k_final(const float* __restrict__ pooled,
                        const float* __restrict__ mlp_w, const float* __restrict__ mlp_b,
                        int n, float* __restrict__ out) {
    int lane = threadIdx.x;
    float logit = 0.f;
    if (lane < 10) {
        float l = mlp_b[lane];
        float invn = 1.0f / (float)n;
        #pragma unroll
        for (int k = 0; k < 10; ++k) l += (pooled[k] * invn) * mlp_w[k * 10 + lane];
        logit = l;
    }
    float lv = (lane < 10) ? logit : -3.0e38f;
    #pragma unroll
    for (int off = 8; off >= 1; off >>= 1) lv = fmaxf(lv, __shfl_xor(lv, off, 16));
    float ev = (lane < 10) ? expf(logit - lv) : 0.f;
    float sv = ev;
    #pragma unroll
    for (int off = 8; off >= 1; off >>= 1) sv += __shfl_xor(sv, off, 16);
    if (lane < 10) out[lane] = ev / sv;
}

// ---------------------------------------------------------------------------
extern "C" void kernel_launch(void* const* d_in, const int* in_sizes, int n_in,
                              void* d_out, int out_size, void* d_ws, size_t ws_size,
                              hipStream_t stream) {
    const float* pos   = (const float*)d_in[0];
    const int*   ei    = (const int*)  d_in[1];
    const float* W1_0e = (const float*)d_in[2];
    const float* W1_1o = (const float*)d_in[3];
    const float* W2_0e = (const float*)d_in[4];
    const float* W2_1o = (const float*)d_in[5];
    const float* mlp_w = (const float*)d_in[6];
    const float* mlp_b = (const float*)d_in[7];
    (void)n_in; (void)ws_size; (void)out_size;

    int n = in_sizes[0] / 3;
    int E = in_sizes[1] / 2;
    const int* src = ei;        // edge_index row 0 (sorted ascending by construction)
    const int* dst = ei + E;    // edge_index row 1

    // Workspace carve-out
    char* ws = (char*)d_ws;
    size_t off = 0;
    auto carve = [&](size_t bytes, size_t align) -> char* {
        off = (off + align - 1) & ~(align - 1);
        char* p = ws + off; off += bytes; return p;
    };
    int*    row_start = (int*)   carve((size_t)(n + 1) * sizeof(int), 16);
    float4* rec       = (float4*)carve((size_t)n * 2 * sizeof(float4), 16);
    float*  a_table   = (float*) carve((size_t)MAXDEG * TROW * sizeof(float), 16);
    float*  c2        = (float*) carve(10 * sizeof(float), 16);
    float*  pooled    = (float*) carve(10 * sizeof(float), 16);

    const int TB = 256;                        // 8 waves per block (K1/K2)
    int grid_rs    = (n + 1 + TB - 1) / TB;
    int grid_sdeg  = (n + 7) / 8;              // one wave32 per node
    int grid_accum = (n + 31) / 32;            // 1024-thread blocks, 32 nodes each

    k_rowstart<<<grid_rs, TB, 0, stream>>>(src, pos, E, n, row_start, rec, pooled);
    k_sdeg    <<<grid_sdeg, TB, 0, stream>>>(rec, dst, row_start, n, rec);
    k_table   <<<1, 512, 0, stream>>>(W1_0e, W2_0e, W1_1o, W2_1o, a_table, c2);
    k_accum   <<<grid_accum, 1024, 0, stream>>>(rec, dst, row_start,
                                                (const float4*)a_table, c2, n, pooled);
    k_final   <<<1, 32, 0, stream>>>(pooled, mlp_w, mlp_b, n, (float*)d_out);
}
